// AVStarTransformerEncoderLayer_59133109731899
// MI455X (gfx1250) — matrix operand order repaired and checked by reading, not compiled, not run on hardware
//
#include <hip/hip_runtime.h>
#include <hip/hip_fp16.h>

typedef _Float16 h16;
typedef __attribute__((ext_vector_type(16))) _Float16 v16h;
typedef __attribute__((ext_vector_type(8)))  float    v8f;

#define BB   8
#define LAA  5
#define DDIM 256
#define HH   8
#define HDD  32
#define FFND 2048
#define LQV  5376              // 64*64+32*32+16*16
#define MV   (BB*LQV)          // 43008
#define MA   (BB*LAA)          // 40

#define GF_GELU    1
#define GF_ACCUM   2
#define GF_WRITEF  4
#define GF_WRITEH  8

// ---------------------------------------------------------------------------
// WMMA GEMM: C[M,N] = act(A[M,K] * B[K,N] + bias (+C))
// A: f16 row-major (lda). B: f16 row-major KxN (ldb); caller guarantees
// every 128-wide column tile is fully in-bounds of the allocation (weights
// are zero-padded to a multiple of 128 columns where needed).
// Block tile 128x128, 8 waves (4x2), wave tile 32x64 (2x4 wmma 16x16x32).
// ---------------------------------------------------------------------------
__global__ __launch_bounds__(256)
void gemm_f16_wmma(const h16* __restrict__ A, int lda,
                   const h16* __restrict__ Bw, int ldb,
                   const float* __restrict__ bias,
                   float* __restrict__ C, h16* __restrict__ Ch, int ldc,
                   int M, int N, int K, int flags)
{
    const int KP = 40;                       // padded LDS stride (halves), 80B rows
    __shared__ __align__(16) h16 As[128 * 40];
    __shared__ __align__(16) h16 Bs[128 * 40];

    const int tid  = threadIdx.x;
    const int lane = tid & 31;
    const int wave = tid >> 5;
    const int lh   = lane & 15;
    const int hi   = lane >> 4;
    const int wm   = wave >> 1;              // 0..3 -> M offset wm*32
    const int wn   = wave & 1;               // 0..1 -> N offset wn*64
    const int blockM = blockIdx.y * 128;
    const int blockN = blockIdx.x * 128;

    v8f acc[2][4];
    const v8f vzero = {0.f,0.f,0.f,0.f,0.f,0.f,0.f,0.f};
    #pragma unroll
    for (int i = 0; i < 2; ++i)
        #pragma unroll
        for (int j = 0; j < 4; ++j) acc[i][j] = vzero;

    union UF { v16h h; unsigned int u[8]; };

    for (int k0 = 0; k0 < K; k0 += 32) {
        // ---- stage A tile: 128x32 halves (512 chunks of 8 halves) ----
        #pragma unroll
        for (int i = 0; i < 2; ++i) {
            int idx = tid + i * 256;         // 0..511
            int row = idx >> 2;
            int kc  = (idx & 3) * 8;
            int grow = blockM + row;
            uint4 v = make_uint4(0u, 0u, 0u, 0u);
            if (grow < M)
                v = *(const uint4*)(A + (size_t)grow * lda + k0 + kc);
            *(uint4*)&As[row * KP + kc] = v;
        }
        // ---- stage B tile transposed: Bs[n][k], 32x128 halves, vector loads ----
        #pragma unroll
        for (int i = 0; i < 2; ++i) {
            int e    = (tid + i * 256) * 8;  // element id, 0..4095
            int krow = e >> 7;               // 0..31
            int ncol = e & 127;              // multiple of 8
            uint4 v = *(const uint4*)(Bw + (size_t)(k0 + krow) * ldb + blockN + ncol);
            const h16* hv = (const h16*)&v;
            #pragma unroll
            for (int j = 0; j < 8; ++j)
                Bs[(ncol + j) * KP + krow] = hv[j];
        }
        __syncthreads();

        // ---- load fragments (CDNA5 16-bit A/B VGPR layouts) ----
        UF af[2], bf[4];
        #pragma unroll
        for (int sm = 0; sm < 2; ++sm) {
            const h16* ap = &As[(wm * 32 + sm * 16 + lh) * KP];
            int kb = hi * 8;
            #pragma unroll
            for (int v = 0; v < 4; ++v)
                af[sm].u[v] = *(const unsigned int*)(ap + kb + 2 * v);
            #pragma unroll
            for (int v = 0; v < 4; ++v)
                af[sm].u[4 + v] = *(const unsigned int*)(ap + 16 + kb + 2 * v);
        }
        #pragma unroll
        for (int sn = 0; sn < 4; ++sn) {
            const h16* bp = &Bs[(wn * 64 + sn * 16 + lh) * KP];
            int kb = hi * 8;
            #pragma unroll
            for (int v = 0; v < 4; ++v)
                bf[sn].u[v] = *(const unsigned int*)(bp + kb + 2 * v);
            #pragma unroll
            for (int v = 0; v < 4; ++v)
                bf[sn].u[4 + v] = *(const unsigned int*)(bp + 16 + kb + 2 * v);
        }
        #pragma unroll
        for (int sm = 0; sm < 2; ++sm)
            #pragma unroll
            for (int sn = 0; sn < 4; ++sn)
                acc[sm][sn] = __builtin_amdgcn_wmma_f32_16x16x32_f16(
                    false, af[sm].h, false, bf[sn].h,
                    (short)0, acc[sm][sn], false, false);
        __syncthreads();
    }

    // ---- epilogue: C layout lane=(N), vgpr r -> M = r + 8*hi ----
    #pragma unroll
    for (int sm = 0; sm < 2; ++sm) {
        #pragma unroll
        for (int sn = 0; sn < 4; ++sn) {
            int gcol = blockN + wn * 64 + sn * 16 + lh;
            #pragma unroll
            for (int r = 0; r < 8; ++r) {
                int grow = blockM + wm * 32 + sm * 16 + r + 8 * hi;
                if (grow < M && gcol < N) {
                    float x = acc[sm][sn][r];
                    if (bias) x += bias[gcol];
                    if (flags & GF_ACCUM)  x += C[(size_t)grow * ldc + gcol];
                    if (flags & GF_GELU)   x = 0.5f * x * (1.f + erff(x * 0.70710678118f));
                    if (flags & GF_WRITEF) C[(size_t)grow * ldc + gcol] = x;
                    if (flags & GF_WRITEH) Ch[(size_t)grow * ldc + gcol] = (h16)x;
                }
            }
        }
    }
}

// ---------------------------------------------------------------------------
__global__ void f32_to_f16_k(const float* __restrict__ x, h16* __restrict__ y, size_t n)
{
    size_t i = (size_t)blockIdx.x * blockDim.x + threadIdx.x;
    if (i < n) y[i] = (h16)x[i];
}

// convert KxN f32 weight into KxLDD f16 with zero padding of columns N..LDD-1
__global__ void cvt_pad_k(const float* __restrict__ src, h16* __restrict__ dst,
                          int K, int N, int ldd)
{
    size_t i = (size_t)blockIdx.x * blockDim.x + threadIdx.x;
    if (i >= (size_t)K * ldd) return;
    int n = (int)(i % ldd);
    int k = (int)(i / ldd);
    dst[i] = (n < N) ? (h16)src[(size_t)k * N + n] : (h16)0.f;
}

__global__ void add_to_f16_k(const float* __restrict__ a, const float* __restrict__ b,
                             h16* __restrict__ y, size_t n)
{
    size_t i = (size_t)blockIdx.x * blockDim.x + threadIdx.x;
    if (i < n) y[i] = (h16)(a[i] + b[i]);
}

// residual + LayerNorm over D=256; one block per row
__global__ __launch_bounds__(256)
void residual_ln_k(const float* __restrict__ x, const float* __restrict__ r,
                   const float* __restrict__ g, const float* __restrict__ b,
                   float* __restrict__ out, h16* __restrict__ outh)
{
    __shared__ float red[256];
    int row = blockIdx.x, t = threadIdx.x;
    size_t base = (size_t)row * 256;
    float v = x[base + t] + r[base + t];
    red[t] = v; __syncthreads();
    for (int s = 128; s > 0; s >>= 1) { if (t < s) red[t] += red[t + s]; __syncthreads(); }
    float mean = red[0] * (1.f / 256.f);
    __syncthreads();
    float d = v - mean;
    red[t] = d * d; __syncthreads();
    for (int s = 128; s > 0; s >>= 1) { if (t < s) red[t] += red[t + s]; __syncthreads(); }
    float var = red[0] * (1.f / 256.f);
    float y = d * rsqrtf(var + 1e-5f) * g[t] + b[t];
    out[base + t] = y;
    if (outh) outh[base + t] = (h16)y;
}

// softmax over 12 deform points, rows = MV*H
__global__ void softmax12_k(float* __restrict__ att, int rows)
{
    int i = blockIdx.x * blockDim.x + threadIdx.x;
    if (i >= rows) return;
    float* p = att + (size_t)i * 12;
    float m = p[0];
    #pragma unroll
    for (int j = 1; j < 12; ++j) m = fmaxf(m, p[j]);
    float s = 0.f;
    float e[12];
    #pragma unroll
    for (int j = 0; j < 12; ++j) { e[j] = __expf(p[j] - m); s += e[j]; }
    float inv = 1.f / s;
    #pragma unroll
    for (int j = 0; j < 12; ++j) p[j] = e[j] * inv;
}

// MSDeform bilinear sampling: thread per (b,q,h,d)
__global__ void msdeform_sample_k(const float* __restrict__ value, // [B,LV,H,HD]
                                  const float* __restrict__ off,   // [B,LQ,H,L,P,2]
                                  const float* __restrict__ att,   // [B,LQ,H,L,P]
                                  const float* __restrict__ ref,   // [B,LQ,L,2]
                                  float* __restrict__ out)         // [B,LQ,256]
{
    size_t idx = (size_t)blockIdx.x * blockDim.x + threadIdx.x;
    if (idx >= (size_t)MV * 256) return;
    int d  = idx & 31;
    int h  = (idx >> 5) & 7;
    size_t bq = idx >> 8;
    int b = (int)(bq / LQV);
    const float* offp = off + bq * 192 + h * 24;
    const float* attp = att + bq * 96  + h * 12;
    const float* refq = ref + bq * 6;
    const int HL[3] = {64, 32, 16}, WL[3] = {64, 32, 16};
    float accv = 0.f;
    int start = 0;
    #pragma unroll
    for (int l = 0; l < 3; ++l) {
        const int Hl = HL[l], Wl = WL[l];
        const float rx = refq[l * 2 + 0], ry = refq[l * 2 + 1];
        const float* vl = value + ((size_t)b * LQV + start) * 256 + h * 32 + d;
        #pragma unroll
        for (int p = 0; p < 4; ++p) {
            float x = (rx + offp[l * 8 + p * 2 + 0] / (float)Wl) * Wl - 0.5f;
            float y = (ry + offp[l * 8 + p * 2 + 1] / (float)Hl) * Hl - 0.5f;
            float fx = floorf(x), fy = floorf(y);
            float dx = x - fx, dy = y - fy;
            int x0 = (int)fx, y0 = (int)fy;
            float s = 0.f;
            #pragma unroll
            for (int cy = 0; cy < 2; ++cy)
                #pragma unroll
                for (int cx = 0; cx < 2; ++cx) {
                    int xi = x0 + cx, yi = y0 + cy;
                    if (xi >= 0 && xi < Wl && yi >= 0 && yi < Hl) {
                        float gv = vl[(size_t)(yi * Wl + xi) * 256];
                        s += gv * (cx ? dx : 1.f - dx) * (cy ? dy : 1.f - dy);
                    }
                }
            accv += s * attp[l * 4 + p];
        }
        start += Hl * Wl;
    }
    out[idx] = accv;
}

// audio self-attention, single head, 5 keys; block per (b,i)
__global__ __launch_bounds__(256)
void audio_self_attn_k(const float* __restrict__ qa, const float* __restrict__ ka,
                       const float* __restrict__ va, float* __restrict__ out)
{
    __shared__ float red[256];
    __shared__ float sc[LAA];
    int b = blockIdx.x / LAA, i = blockIdx.x % LAA, t = threadIdx.x;
    float qv = qa[((size_t)b * LAA + i) * 256 + t];
    for (int j = 0; j < LAA; ++j) {
        red[t] = qv * ka[((size_t)b * LAA + j) * 256 + t];
        __syncthreads();
        for (int s = 128; s > 0; s >>= 1) { if (t < s) red[t] += red[t + s]; __syncthreads(); }
        if (t == 0) sc[j] = red[0] * 0.0625f;   // 256^-0.5
        __syncthreads();
    }
    if (t == 0) {
        float m = sc[0];
        for (int j = 1; j < LAA; ++j) m = fmaxf(m, sc[j]);
        float s = 0.f;
        for (int j = 0; j < LAA; ++j) { sc[j] = __expf(sc[j] - m); s += sc[j]; }
        for (int j = 0; j < LAA; ++j) sc[j] /= s;
    }
    __syncthreads();
    float o = 0.f;
    for (int j = 0; j < LAA; ++j) o += sc[j] * va[((size_t)b * LAA + j) * 256 + t];
    out[((size_t)b * LAA + i) * 256 + t] = o;
}

// 8-head cross attention, 5 keys: thread per (b,q,h)
__global__ void cross_attn_k(const float* __restrict__ qv, const float* __restrict__ ka,
                             const float* __restrict__ va, float* __restrict__ out)
{
    size_t i = (size_t)blockIdx.x * blockDim.x + threadIdx.x;
    if (i >= (size_t)MV * HH) return;
    int h = i & 7;
    size_t bq = i >> 3;
    int b = (int)(bq / LQV);
    const float* qp = qv + bq * 256 + h * 32;
    float sc[LAA];
    float m = -1e30f;
    for (int j = 0; j < LAA; ++j) {
        const float* kp = ka + ((size_t)b * LAA + j) * 256 + h * 32;
        float s = 0.f;
        #pragma unroll
        for (int d = 0; d < 32; ++d) s += qp[d] * kp[d];
        s *= 0.17677669529f;                // 32^-0.5
        sc[j] = s; m = fmaxf(m, s);
    }
    float sum = 0.f;
    for (int j = 0; j < LAA; ++j) { sc[j] = __expf(sc[j] - m); sum += sc[j]; }
    float inv = 1.f / sum;
    float* op = out + bq * 256 + h * 32;
    #pragma unroll
    for (int d = 0; d < 32; ++d) {
        float o = 0.f;
        for (int j = 0; j < LAA; ++j)
            o += sc[j] * va[((size_t)b * LAA + j) * 256 + h * 32 + d];
        op[d] = o * inv;
    }
}

// ---------------------------------------------------------------------------
extern "C" void kernel_launch(void* const* d_in, const int* in_sizes, int n_in,
                              void* d_out, int out_size, void* d_ws, size_t ws_size,
                              hipStream_t stream)
{
    (void)in_sizes; (void)n_in; (void)out_size; (void)ws_size;
    const float* src_a  = (const float*)d_in[0];
    const float* src_v  = (const float*)d_in[1];
    const float* pos    = (const float*)d_in[2];
    const float* refp   = (const float*)d_in[3];
    const float* W_val  = (const float*)d_in[4];  const float* b_val = (const float*)d_in[5];
    const float* W_off  = (const float*)d_in[6];  const float* b_off = (const float*)d_in[7];
    const float* W_att  = (const float*)d_in[8];  const float* b_att = (const float*)d_in[9];
    const float* W_out  = (const float*)d_in[10]; const float* b_out = (const float*)d_in[11];
    const float* ln_v_g = (const float*)d_in[12]; const float* ln_v_b = (const float*)d_in[13];
    const float* Wq_a   = (const float*)d_in[14]; const float* bq_a  = (const float*)d_in[15];
    const float* Wk_a   = (const float*)d_in[16]; const float* bk_a  = (const float*)d_in[17];
    const float* Wv_a   = (const float*)d_in[18]; const float* bv_a  = (const float*)d_in[19];
    const float* Wo_a   = (const float*)d_in[20]; const float* bo_a  = (const float*)d_in[21];
    const float* ln_a_g = (const float*)d_in[22]; const float* ln_a_b = (const float*)d_in[23];
    const float* Wq_va  = (const float*)d_in[24]; const float* bq_va = (const float*)d_in[25];
    const float* Wk_va  = (const float*)d_in[26]; const float* bk_va = (const float*)d_in[27];
    const float* Wv_va  = (const float*)d_in[28]; const float* bv_va = (const float*)d_in[29];
    const float* Wo_va  = (const float*)d_in[30]; const float* bo_va = (const float*)d_in[31];
    const float* ln_va_g= (const float*)d_in[32]; const float* ln_va_b=(const float*)d_in[33];
    const float* W1_v   = (const float*)d_in[34]; const float* b1_v  = (const float*)d_in[35];
    const float* W2_v   = (const float*)d_in[36]; const float* b2_v  = (const float*)d_in[37];
    const float* ln_vf_g= (const float*)d_in[38]; const float* ln_vf_b=(const float*)d_in[39];
    const float* W1_a   = (const float*)d_in[40]; const float* b1_a  = (const float*)d_in[41];
    const float* W2_a   = (const float*)d_in[42]; const float* b2_a  = (const float*)d_in[43];
    const float* ln_af_g= (const float*)d_in[44]; const float* ln_af_b=(const float*)d_in[45];

    // ---- workspace arena ----
    char* ws = (char*)d_ws;
    size_t cur = 0;
    auto alloc = [&](size_t bytes) -> void* {
        void* p = ws + cur;
        cur = (cur + bytes + 255) & ~(size_t)255;
        return p;
    };
    const size_t D2 = 256 * 256;
    h16* whval = (h16*)alloc(D2 * 2);
    h16* whoff = (h16*)alloc(D2 * 2);               // padded 192 -> 256 cols
    h16* whatt = (h16*)alloc((size_t)256 * 128 * 2); // padded 96 -> 128 cols
    h16* whout = (h16*)alloc(D2 * 2);
    h16* whqa  = (h16*)alloc(D2 * 2);
    h16* whka  = (h16*)alloc(D2 * 2);
    h16* whva  = (h16*)alloc(D2 * 2);
    h16* whoa  = (h16*)alloc(D2 * 2);
    h16* whqva = (h16*)alloc(D2 * 2);
    h16* whkva = (h16*)alloc(D2 * 2);
    h16* whvva = (h16*)alloc(D2 * 2);
    h16* whova = (h16*)alloc(D2 * 2);
    h16* wh1v  = (h16*)alloc((size_t)256 * 2048 * 2);
    h16* wh2v  = (h16*)alloc((size_t)2048 * 256 * 2);
    h16* wh1a  = (h16*)alloc((size_t)256 * 2048 * 2);
    h16* wh2a  = (h16*)alloc((size_t)2048 * 256 * 2);

    h16*   acth = (h16*)alloc((size_t)MV * 256 * 2);   // vh / qh / samp_h / co_h
    h16*   svh  = (h16*)alloc((size_t)MV * 256 * 2);
    float* bufA = (float*)alloc((size_t)MV * 256 * 4); // value / v2 / v3p
    char*  bufB = (char*)alloc((size_t)MV * 288 * 4);  // off+att / co / ffn h1(half)
    float* bufC = (float*)alloc((size_t)MV * 256 * 4); // sampled / qv / ffn-out y
    float* bufD = (float*)alloc((size_t)MV * 256 * 4); // sv (running)

    h16*   srcah = (h16*)alloc((size_t)MA * 256 * 2);
    h16*   aoh   = (h16*)alloc((size_t)MA * 256 * 2);
    h16*   sah   = (h16*)alloc((size_t)MA * 256 * 2);
    h16*   ah1   = (h16*)alloc((size_t)MA * 2048 * 2);
    float* aqa = (float*)alloc((size_t)MA * 256 * 4);
    float* aka = (float*)alloc((size_t)MA * 256 * 4);
    float* ava = (float*)alloc((size_t)MA * 256 * 4);
    float* aao = (float*)alloc((size_t)MA * 256 * 4);
    float* aa2 = (float*)alloc((size_t)MA * 256 * 4);
    float* asa = (float*)alloc((size_t)MA * 256 * 4);
    float* akv = (float*)alloc((size_t)MA * 256 * 4);
    float* avv = (float*)alloc((size_t)MA * 256 * 4);
    float* aya = (float*)alloc((size_t)MA * 256 * 4);

    float* offb = (float*)bufB;
    float* attb = (float*)(bufB + (size_t)MV * 192 * 4);
    float* cob  = (float*)bufB;
    h16*   h1b  = (h16*)bufB;

    float* out_sa = (float*)d_out;                 // [B,LA,D]
    float* out_sv = out_sa + (size_t)MA * 256;     // [B,LV,D]

    auto cvt = [&](const float* s, h16* d, size_t n) {
        f32_to_f16_k<<<(unsigned)((n + 255) / 256), 256, 0, stream>>>(s, d, n);
    };
    auto cvtpad = [&](const float* s, h16* d, int K, int N, int ldd) {
        size_t n = (size_t)K * ldd;
        cvt_pad_k<<<(unsigned)((n + 255) / 256), 256, 0, stream>>>(s, d, K, N, ldd);
    };
    auto gemm = [&](const h16* A, int lda, const h16* Bm, int ldb, const float* bias,
                    float* C, h16* Ch, int ldc, int M, int N, int K, int flags) {
        dim3 g((N + 127) / 128, (M + 127) / 128);
        gemm_f16_wmma<<<g, 256, 0, stream>>>(A, lda, Bm, ldb, bias, C, Ch, ldc, M, N, K, flags);
    };
    auto ln = [&](const float* x, const float* r, const float* g, const float* b,
                  float* o, h16* oh, int rows) {
        residual_ln_k<<<rows, 256, 0, stream>>>(x, r, g, b, o, oh);
    };

    // ---- weight conversions to f16 (pad non-multiple-of-128 widths) ----
    cvt(W_val, whval, D2);
    cvtpad(W_off, whoff, 256, 192, 256);
    cvtpad(W_att, whatt, 256, 96, 128);
    cvt(W_out, whout, D2);
    cvt(Wq_a, whqa, D2);  cvt(Wk_a, whka, D2);  cvt(Wv_a, whva, D2);  cvt(Wo_a, whoa, D2);
    cvt(Wq_va, whqva, D2); cvt(Wk_va, whkva, D2); cvt(Wv_va, whvva, D2); cvt(Wo_va, whova, D2);
    cvt(W1_v, wh1v, (size_t)256 * 2048); cvt(W2_v, wh2v, (size_t)2048 * 256);
    cvt(W1_a, wh1a, (size_t)256 * 2048); cvt(W2_a, wh2a, (size_t)2048 * 256);
    cvt(src_a, srcah, (size_t)MA * 256);

    // ================= visual: MSDeformAttn =================
    cvt(src_v, acth, (size_t)MV * 256);                                  // vh
    gemm(acth, 256, whval, 256, b_val, bufA, nullptr, 256, MV, 256, 256, GF_WRITEF); // value
    {
        size_t n = (size_t)MV * 256;
        add_to_f16_k<<<(unsigned)((n + 255) / 256), 256, 0, stream>>>(src_v, pos, acth, n); // qh
    }
    gemm(acth, 256, whoff, 256, b_off, offb, nullptr, 192, MV, 192, 256, GF_WRITEF);
    gemm(acth, 256, whatt, 128, b_att, attb, nullptr, 96,  MV, 96,  256, GF_WRITEF);
    softmax12_k<<<(MV * HH + 255) / 256, 256, 0, stream>>>(attb, MV * HH);
    {
        size_t n = (size_t)MV * 256;
        msdeform_sample_k<<<(unsigned)((n + 255) / 256), 256, 0, stream>>>(bufA, offb, attb, refp, bufC);
    }
    cvt(bufC, acth, (size_t)MV * 256);                                   // samp_h
    gemm(acth, 256, whout, 256, b_out, bufA, nullptr, 256, MV, 256, 256, GF_WRITEF); // v2
    ln(src_v, bufA, ln_v_g, ln_v_b, bufD, svh, MV);                      // sv, svh

    // ================= audio self-attention =================
    gemm(srcah, 256, whqa, 256, bq_a, aqa, nullptr, 256, MA, 256, 256, GF_WRITEF);
    gemm(srcah, 256, whka, 256, bk_a, aka, nullptr, 256, MA, 256, 256, GF_WRITEF);
    gemm(srcah, 256, whva, 256, bv_a, ava, nullptr, 256, MA, 256, 256, GF_WRITEF);
    audio_self_attn_k<<<MA, 256, 0, stream>>>(aqa, aka, ava, aao);
    cvt(aao, aoh, (size_t)MA * 256);
    gemm(aoh, 256, whoa, 256, bo_a, aa2, nullptr, 256, MA, 256, 256, GF_WRITEF);
    ln(src_a, aa2, ln_a_g, ln_a_b, asa, sah, MA);                        // sa, sah

    // ================= cross attention (sv <- sa) =================
    gemm(sah, 256, whkva, 256, bk_va, akv, nullptr, 256, MA, 256, 256, GF_WRITEF);
    gemm(sah, 256, whvva, 256, bv_va, avv, nullptr, 256, MA, 256, 256, GF_WRITEF);
    gemm(svh, 256, whqva, 256, bq_va, bufC, nullptr, 256, MV, 256, 256, GF_WRITEF); // qv
    cross_attn_k<<<(MV * HH + 255) / 256, 256, 0, stream>>>(bufC, akv, avv, cob);
    cvt(cob, acth, (size_t)MV * 256);                                    // co_h
    gemm(acth, 256, whova, 256, bo_va, bufA, nullptr, 256, MV, 256, 256, GF_WRITEF); // v3
    ln(bufD, bufA, ln_va_g, ln_va_b, bufD, svh, MV);                     // sv, svh

    // ================= visual FFN (K-chunked: 4 x 512) =================
    for (int c = 0; c < 4; ++c) {
        gemm(svh, 256, wh1v + c * 512, 2048, b1_v + c * 512,
             nullptr, h1b, 512, MV, 512, 256, GF_GELU | GF_WRITEH);
        gemm(h1b, 512, wh2v + (size_t)c * 512 * 256, 256,
             (c == 0 ? b2_v : nullptr), bufC, nullptr, 256, MV, 256, 512,
             GF_WRITEF | (c ? GF_ACCUM : 0));
    }
    ln(bufD, bufC, ln_vf_g, ln_vf_b, out_sv, nullptr, MV);               // final sv

    // ================= audio FFN =================
    gemm(sah, 256, wh1a, 2048, b1_a, nullptr, ah1, 2048, MA, 2048, 256, GF_GELU | GF_WRITEH);
    gemm(ah1, 2048, wh2a, 256, b2_a, aya, nullptr, 256, MA, 256, 2048, GF_WRITEF);
    ln(asa, aya, ln_af_g, ln_af_b, out_sa, nullptr, MA);                 // final sa
}
